// FasterRCNN_48670569398570
// MI455X (gfx1250) — compile-verified, matrix operand order checked by reading, not run on hardware
//
#include <hip/hip_runtime.h>
#include <hip/hip_bf16.h>
#include <math.h>

typedef __attribute__((ext_vector_type(16))) _Float16 v16h;
typedef __attribute__((ext_vector_type(8)))  float    v8f;
typedef __attribute__((ext_vector_type(4)))  unsigned int u32x4;
typedef __attribute__((ext_vector_type(8)))  int          i32x8;
typedef __attribute__((ext_vector_type(4)))  int          i32x4;

#define IMG_SZ   512
#define FHH      32
#define FWW      32
#define A_ANCH   9
#define NANCH    (FHH*FWW*A_ANCH)   // 9216
#define KTOP     300
#define OHP      7
#define OWP      7
#define CCH      512
#define NCLS     21
#define BN_EPS   1e-5f
#define NMS_T    0.7f
#define MIN_BOX  16.0f

// ---- WMMA fragment K-index mapping (CDNA5 16-bit A/B layout, ISA 7.12.2) ----
// lane l: row/col = l&15, half = l>>4 ; pair p (vgpr) covers K = kbase, kbase+1
// Per lane the 16 halves cover k in [8h, 8h+8) U [16+8h, 24+8h): two contiguous
// 8-float runs -> four 16-byte loads per fragment.
__device__ __forceinline__ int kpair_base(int p, int half) {
  return ((p & 4) ? (16 + ((p & 3) << 1)) : (p << 1)) + (half << 3);
}

// Pack 16 consecutive-k f32 values (two 8-float runs) into a v16h fragment.
__device__ __forceinline__ v16h pack_frag(float4 q0, float4 q1, float4 q2, float4 q3) {
  v16h f;
  f[0]  = (_Float16)q0.x; f[1]  = (_Float16)q0.y; f[2]  = (_Float16)q0.z; f[3]  = (_Float16)q0.w;
  f[4]  = (_Float16)q1.x; f[5]  = (_Float16)q1.y; f[6]  = (_Float16)q1.z; f[7]  = (_Float16)q1.w;
  f[8]  = (_Float16)q2.x; f[9]  = (_Float16)q2.y; f[10] = (_Float16)q2.z; f[11] = (_Float16)q2.w;
  f[12] = (_Float16)q3.x; f[13] = (_Float16)q3.y; f[14] = (_Float16)q3.z; f[15] = (_Float16)q3.w;
  return f;
}

// ---- Tensor Data Mover: async copy of a 2-D f32 tile (32 wide x 32 rows)
// from global (row stride = `stride` floats) into LDS at lds_off.
// D# packing per CDNA5 ISA §8.3/§8.4. 6-arg builtin on this toolchain.
__device__ __forceinline__ void tdm_load_b_panel(const float* gaddr, unsigned lds_off,
                                                 int kdim_rem, int rows_rem, int stride) {
  unsigned long long ga = (unsigned long long)(size_t)gaddr;
  u32x4 g0;
  g0[0] = 1u;                                            // count=1, user descriptor
  g0[1] = lds_off;                                       // lds_addr (bytes)
  g0[2] = (unsigned)(ga & 0xFFFFFFFFull);                // global_addr[31:0]
  g0[3] = (unsigned)((ga >> 32) & 0x1FFFFFFull) | (2u << 30);  // addr[56:32] | type=2
  i32x8 g1;
  unsigned td0 = (unsigned)kdim_rem;                     // tensor_dim0
  unsigned td1 = (unsigned)rows_rem;                     // tensor_dim1
  g1[0] = (int)(2u << 16);                               // wg_mask=0, data_size=4B
  g1[1] = (int)((td0 & 0xFFFFu) << 16);                  // abar=0 | tensor_dim0[15:0]
  g1[2] = (int)((td0 >> 16) | ((td1 & 0xFFFFu) << 16));  // td0[31:16] | td1[15:0]
  g1[3] = (int)((td1 >> 16) | (32u << 16));              // td1[31:16] | tile_dim0=32
  g1[4] = (int)32u;                                      // tile_dim1=32, tile_dim2=0
  g1[5] = (int)stride;                                   // tensor_dim0_stride[31:0]
  g1[6] = 0;                                             // stride hi | dim1_stride lo
  g1[7] = 0;
  i32x4 z4 = {0, 0, 0, 0};
  i32x8 z8 = {0, 0, 0, 0, 0, 0, 0, 0};
  __builtin_amdgcn_tensor_load_to_lds(g0, g1, z4, z4, z8, 0);
}

// =====================================================================
// Implicit-GEMM convolution, templated on filter/output geometry so all
// index math lowers to mul-shift. One wave -> 16x32 (spatial x Cout) tile,
// two WMMAs per K-step sharing the A fragment. Branch-free inner loop:
// OOR rows/cols are index-clamped (never stored); A-side k/pad zeroing is
// done with selects, which also nullifies any clamped B garbage.
// USE_TDM: double-buffered LDS staging of the 32x32 f32 weight panel via
// the Tensor Data Mover (requires Kdim % 32 == 0).
// =====================================================================
template <int KH, int KW, int OWt, bool USE_TDM>
__global__ void __launch_bounds__(32)
conv_wmma_kernel(const float* __restrict__ in, const float* __restrict__ wgt,
                 const float* __restrict__ bias,
                 const float* __restrict__ bn_g, const float* __restrict__ bn_b,
                 const float* __restrict__ bn_m, const float* __restrict__ bn_v,
                 float* __restrict__ out,
                 int Cin, int H, int W, int Cout, int OHh,
                 int stride, int pad, int relu)
{
  __shared__ float bstage[2][32 * 32];   // double-buffered TDM staging
  const int lane = threadIdx.x & 31;
  const int half = lane >> 4;
  const int l16  = lane & 15;
  const int M    = OHh * OWt;
  const int Kdim = Cin * KH * KW;
  constexpr int khw = KH * KW;
  const int m0 = blockIdx.x * 16;
  const int n0 = blockIdx.y * 32;
  const float* inB  = in  + (size_t)blockIdx.z * Cin  * H * W;
  float*       outB = out + (size_t)blockIdx.z * Cout * M;

  const int am   = m0 + l16;             // A row this lane supplies
  const int bn0  = n0 + l16;             // B col, first N-tile
  const int bn1  = n0 + 16 + l16;        // B col, second N-tile
  const int amc  = min(am, M - 1);       // clamped (garbage rows never stored)
  const bool mok = (am < M);
  const int oy  = amc / OWt, ox = amc - oy * OWt;
  const int iy0 = oy * stride - pad;
  const int ix0 = ox * stride - pad;
  const int CHW = Cin * H * W;

  const unsigned lds0 = (unsigned)(size_t)&bstage[0][0];
  const unsigned lds1 = (unsigned)(size_t)&bstage[1][0];

  const int bn0c = min(bn0, Cout - 1);
  const int bn1c = min(bn1, Cout - 1);
  const float* w0 = wgt + (size_t)bn0c * Kdim;   // direct-load path rows
  const float* w1 = wgt + (size_t)bn1c * Kdim;

  const int nch = (Kdim + 31) >> 5;
  if (USE_TDM)
    tdm_load_b_panel(wgt + (size_t)n0 * Kdim, lds0, Kdim, Cout - n0, Kdim);

  v8f acc0 = {}, acc1 = {};
  for (int c = 0; c < nch; ++c) {
    const int k0 = c << 5;
    if (USE_TDM) {
      if (c + 1 < nch) {
        // prior reads of the buffer being overwritten have long retired,
        // but make it explicit before the TDM write, then keep 1 in flight.
        asm volatile("s_wait_dscnt 0" ::: "memory");
        tdm_load_b_panel(wgt + (size_t)n0 * Kdim + (k0 + 32),
                         ((c + 1) & 1) ? lds1 : lds0,
                         Kdim - k0 - 32, Cout - n0, Kdim);
        __builtin_amdgcn_s_wait_tensorcnt(1);   // chunk c complete (in-order)
      } else {
        __builtin_amdgcn_s_wait_tensorcnt(0);
      }
    }

    // ---- A fragment: branch-free im2col gather ----
    v16h af;
#pragma unroll
    for (int p = 0; p < 8; ++p) {
      const int kb = kpair_base(p, half);
#pragma unroll
      for (int e = 0; e < 2; ++e) {
        const int k = k0 + kb + e;
        int ci = k / khw; int rr = k - ci * khw;
        int ky = rr / KW; int kx = rr - ky * KW;
        int iy = iy0 + ky, ix = ix0 + kx;
        bool ok = mok && (k < Kdim) &&
                  ((unsigned)iy < (unsigned)H) && ((unsigned)ix < (unsigned)W);
        int idx = ci * H * W + iy * W + ix;
        idx = min(max(idx, 0), CHW - 1);
        float raw = inB[idx];
        af[2 * p + e] = (_Float16)(ok ? raw : 0.f);
      }
    }

    // ---- B fragments ----
    v16h bf0, bf1;
    if (USE_TDM) {
      const float* bst = &bstage[c & 1][0];
      const float4* r0q = (const float4*)(bst + l16 * 32 + 8 * half);
      const float4* r1q = (const float4*)(bst + (16 + l16) * 32 + 8 * half);
      bf0 = pack_frag(r0q[0], r0q[1], r0q[4], r0q[5]);   // +0,+4,+16,+20 floats
      bf1 = pack_frag(r1q[0], r1q[1], r1q[4], r1q[5]);
    } else {
      // clamped-k loads; A-side zeros make k>=Kdim contributions vanish
#pragma unroll
      for (int p = 0; p < 8; ++p) {
        const int kb = kpair_base(p, half);
#pragma unroll
        for (int e = 0; e < 2; ++e) {
          int k = min(k0 + kb + e, Kdim - 1);
          bf0[2 * p + e] = (_Float16)w0[k];
          bf1[2 * p + e] = (_Float16)w1[k];
        }
      }
    }

    acc0 = __builtin_amdgcn_wmma_f32_16x16x32_f16(false, af, false, bf0,
                                                  (short)0, acc0, false, false);
    acc1 = __builtin_amdgcn_wmma_f32_16x16x32_f16(false, af, false, bf1,
                                                  (short)0, acc1, false, false);
  }

#pragma unroll
  for (int t = 0; t < 2; ++t) {
    const int bn = (t == 0) ? bn0 : bn1;
    const v8f& acc = (t == 0) ? acc0 : acc1;
    if (bn < Cout) {
      float b   = bias ? bias[bn] : 0.f;
      float scl = 1.f, sh = 0.f;
      if (bn_g) {
        float inv = bn_g[bn] * rsqrtf(bn_v[bn] + BN_EPS);
        scl = inv; sh = bn_b[bn] - bn_m[bn] * inv;
      }
#pragma unroll
      for (int r = 0; r < 8; ++r) {
        int m = m0 + r + 8 * half;
        if (m < M) {
          float v = (acc[r] + b) * scl + sh;
          if (relu) v = fmaxf(v, 0.f);
          outB[(size_t)bn * M + m] = v;
        }
      }
    }
  }
}

// =====================================================================
// Dense GEMM for FC layers: out[m][n] = A[m][:] . W[n][:] + bias[n]
// One wave -> 16x32 tile, 2 WMMAs/K-step. Branch-free: rows/cols index-
// clamped, all operand loads are b128, prefetch 4 chunks ahead.
// K % 32 == 0 (25088, 4096).
// =====================================================================
__global__ void __launch_bounds__(32)
gemm_wmma_kernel(const float* __restrict__ A, const float* __restrict__ Wt,
                 const float* __restrict__ bias, float* __restrict__ out,
                 int M, int N, int Kdim, int relu)
{
  const int lane = threadIdx.x & 31;
  const int half = lane >> 4;
  const int l16  = lane & 15;
  const int m0 = blockIdx.x * 16, n0 = blockIdx.y * 32;
  const int am  = m0 + l16;
  const int bn0 = n0 + l16;
  const int bn1 = n0 + 16 + l16;
  const int amc  = min(am,  M - 1);
  const int bn0c = min(bn0, N - 1);
  const int bn1c = min(bn1, N - 1);
  const float* arow = A  + (size_t)amc  * Kdim + 8 * half;
  const float* b0r  = Wt + (size_t)bn0c * Kdim + 8 * half;
  const float* b1r  = Wt + (size_t)bn1c * Kdim + 8 * half;

  v8f acc0 = {}, acc1 = {};
  for (int k0 = 0; k0 < Kdim; k0 += 32) {
    if (k0 + 128 < Kdim) {            // prefetch 4 chunks ahead
      __builtin_prefetch(arow + k0 + 128, 0, 1);
      __builtin_prefetch(b0r  + k0 + 128, 0, 1);
      __builtin_prefetch(b1r  + k0 + 128, 0, 1);
    }
    const float4* qa = (const float4*)(arow + k0);
    const float4* q0 = (const float4*)(b0r  + k0);
    const float4* q1 = (const float4*)(b1r  + k0);
    v16h af  = pack_frag(qa[0], qa[1], qa[4], qa[5]);
    v16h bf0 = pack_frag(q0[0], q0[1], q0[4], q0[5]);
    v16h bf1 = pack_frag(q1[0], q1[1], q1[4], q1[5]);
    acc0 = __builtin_amdgcn_wmma_f32_16x16x32_f16(false, af, false, bf0,
                                                  (short)0, acc0, false, false);
    acc1 = __builtin_amdgcn_wmma_f32_16x16x32_f16(false, af, false, bf1,
                                                  (short)0, acc1, false, false);
  }

#pragma unroll
  for (int t = 0; t < 2; ++t) {
    const int bn = (t == 0) ? bn0 : bn1;
    const v8f& acc = (t == 0) ? acc0 : acc1;
    if (bn < N) {
      float b = bias ? bias[bn] : 0.f;
#pragma unroll
      for (int r = 0; r < 8; ++r) {
        int m = m0 + r + 8 * half;
        if (m < M) {
          float v = acc[r] + b;
          if (relu) v = fmaxf(v, 0.f);
          out[(size_t)m * N + bn] = v;
        }
      }
    }
  }
}

// =====================================================================
// RPN post: reg channels -> per-anchor rows; prob = sigmoid(c1-c0);
// masked = prob>0.5 ? prob : -1.  r1 layout: [54][FH*FW] (NCHW).
// =====================================================================
__global__ void rpn_post_kernel(const float* __restrict__ r1,
                                float* __restrict__ reg,
                                float* __restrict__ prob,
                                float* __restrict__ masked)
{
  int gid = blockIdx.x * blockDim.x + threadIdx.x;
  if (gid >= NANCH) return;
  int a = gid % A_ANCH;
  int m = gid / A_ANCH;
  const int MM = FHH * FWW;
#pragma unroll
  for (int c = 0; c < 4; ++c)
    reg[gid * 4 + c] = r1[(4 * a + c) * MM + m];
  float c0 = r1[(4 * A_ANCH + 2 * a)     * MM + m];
  float c1 = r1[(4 * A_ANCH + 2 * a + 1) * MM + m];
  float pr = 1.f / (1.f + __expf(c0 - c1));
  prob[gid]   = pr;
  masked[gid] = (pr > 0.5f) ? pr : -1.f;
}

// Iterative top-K (descending, ties -> lower index). Destroys `masked`
// (rebuilt each call by rpn_post).
__global__ void __launch_bounds__(512)
topk_kernel(float* __restrict__ masked, float* __restrict__ scores,
            int* __restrict__ idx)
{
  __shared__ float sv[512];
  __shared__ int   si[512];
  const int tid = threadIdx.x;
  for (int r = 0; r < KTOP; ++r) {
    float bv = -1e30f; int bi = NANCH;
    for (int i = tid; i < NANCH; i += 512) {
      float v = masked[i];
      if (v > bv || (v == bv && i < bi)) { bv = v; bi = i; }
    }
    sv[tid] = bv; si[tid] = bi;
    __syncthreads();
    for (int s = 256; s > 0; s >>= 1) {
      if (tid < s) {
        float ov = sv[tid + s]; int oi = si[tid + s];
        if (ov > sv[tid] || (ov == sv[tid] && oi < si[tid])) { sv[tid] = ov; si[tid] = oi; }
      }
      __syncthreads();
    }
    if (tid == 0) {
      scores[r] = sv[0];
      idx[r]    = si[0];
      masked[si[0]] = -2e30f;
    }
    __syncthreads();
  }
}

__global__ void decode_kernel(const float* __restrict__ anchors,
                              const float* __restrict__ reg,
                              const float* __restrict__ scores,
                              const int* __restrict__ idx,
                              float* __restrict__ boxes, int* __restrict__ valid)
{
  int t = blockIdx.x * blockDim.x + threadIdx.x;
  if (t >= KTOP) return;
  int id = idx[t];
  float ax1 = anchors[id*4+0], ay1 = anchors[id*4+1];
  float ax2 = anchors[id*4+2], ay2 = anchors[id*4+3];
  float w = ax2 - ax1, h = ay2 - ay1;
  float cx = ax1 + 0.5f * w, cy = ay1 + 0.5f * h;
  float dx = reg[id*4+0], dy = reg[id*4+1], dw = reg[id*4+2], dh = reg[id*4+3];
  float ncx = cx + dx * w, ncy = cy + dy * h;
  float nw = w * __expf(dw), nh = h * __expf(dh);
  float x1 = ncx - 0.5f * nw, y1 = ncy - 0.5f * nh;
  float x2 = ncx + 0.5f * nw, y2 = ncy + 0.5f * nh;
  x1 = fminf(fmaxf(x1, 0.f), (float)IMG_SZ);
  y1 = fminf(fmaxf(y1, 0.f), (float)IMG_SZ);
  x2 = fminf(fmaxf(x2, 0.f), (float)IMG_SZ);
  y2 = fminf(fmaxf(y2, 0.f), (float)IMG_SZ);
  boxes[t*4+0] = x1; boxes[t*4+1] = y1; boxes[t*4+2] = x2; boxes[t*4+3] = y2;
  valid[t] = (scores[t] > 0.5f) && (x2 - x1 >= MIN_BOX) && (y2 - y1 >= MIN_BOX);
}

__global__ void iou_kernel(const float* __restrict__ boxes, float* __restrict__ ious)
{
  int t = blockIdx.x * blockDim.x + threadIdx.x;
  if (t >= KTOP * KTOP) return;
  int i = t / KTOP, j = t - (t / KTOP) * KTOP;
  float ix1 = boxes[i*4+0], iy1 = boxes[i*4+1], ix2 = boxes[i*4+2], iy2 = boxes[i*4+3];
  float jx1 = boxes[j*4+0], jy1 = boxes[j*4+1], jx2 = boxes[j*4+2], jy2 = boxes[j*4+3];
  float ai = (ix2 - ix1) * (iy2 - iy1);
  float aj = (jx2 - jx1) * (jy2 - jy1);
  float lx = fmaxf(ix1, jx1), ly = fmaxf(iy1, jy1);
  float rx = fminf(ix2, jx2), ry = fminf(iy2, jy2);
  float iw = fmaxf(rx - lx, 0.f), ih = fmaxf(ry - ly, 0.f);
  float inter = iw * ih;
  ious[t] = inter / (ai + aj - inter + 1e-9f);
}

// Sequential greedy NMS: keep[i] = valid[i] && !any(keep[j<i] && iou[j][i]>T)
__global__ void __launch_bounds__(512)
nms_kernel(const float* __restrict__ ious, const int* __restrict__ valid,
           int* __restrict__ keep)
{
  __shared__ int skeep[KTOP];
  __shared__ int ssup;
  int tid = threadIdx.x;
  if (tid < KTOP) skeep[tid] = 0;
  __syncthreads();
  for (int i = 0; i < KTOP; ++i) {
    if (tid == 0) ssup = 0;
    __syncthreads();
    if (tid < i && tid < KTOP && skeep[tid] && ious[tid * KTOP + i] > NMS_T)
      atomicOr(&ssup, 1);
    __syncthreads();
    if (tid == 0) skeep[i] = valid[i] && !ssup;
    __syncthreads();
  }
  if (tid < KTOP) keep[tid] = skeep[tid];
}

// Per-channel 2D prefix sum with zero-padded first row/col: II[C][33][33]
__global__ void __launch_bounds__(64)
integral_kernel(const float* __restrict__ f, float* __restrict__ II)
{
  __shared__ float tile[33 * 33];
  const int c = blockIdx.x, t = threadIdx.x;
  for (int i = t; i < 33; i += 64) { tile[i] = 0.f; tile[i * 33] = 0.f; }
  __syncthreads();
  if (t < 32) {
    float run = 0.f;
    for (int x = 0; x < 32; ++x) {
      run += f[((size_t)c * 32 + t) * 32 + x];
      tile[(t + 1) * 33 + (x + 1)] = run;
    }
  }
  __syncthreads();
  if (t < 32) {
    float run = 0.f;
    int x = t + 1;
    for (int y = 1; y < 33; ++y) { run += tile[y * 33 + x]; tile[y * 33 + x] = run; }
  }
  __syncthreads();
  for (int i = t; i < 33 * 33; i += 64) II[(size_t)c * 1089 + i] = tile[i];
}

// ROI average pooling (7x7) via integral image -> rois[K][512][7][7]
__global__ void __launch_bounds__(256)
roipool_kernel(const float* __restrict__ II, const float* __restrict__ boxes,
               float* __restrict__ rois)
{
  __shared__ int r0[OHP], r1[OHP], c0[OWP], c1[OWP];
  const int k = blockIdx.x, t = threadIdx.x;
  if (t == 0) {
    const float s = (float)FWW / (float)IMG_SZ;
    int fx1 = (int)floorf(boxes[k*4+0] * s);
    int fy1 = (int)floorf(boxes[k*4+1] * s);
    int fx2 = (int)floorf(boxes[k*4+2] * s) + 1;
    int fy2 = (int)floorf(boxes[k*4+3] * s) + 1;
    int x1 = min(max(fx1, 0), FWW - 1);
    int y1 = min(max(fy1, 0), FHH - 1);
    int x2 = min(max(fx2, x1 + 1), FWW);
    int y2 = min(max(fy2, y1 + 1), FHH);
    int sh = y2 - y1, sw = x2 - x1;
    for (int i = 0; i < OHP; ++i) {
      r0[i] = y1 + (i * sh) / OHP;
      r1[i] = y1 + ((i + 1) * sh + OHP - 1) / OHP;
    }
    for (int j = 0; j < OWP; ++j) {
      c0[j] = x1 + (j * sw) / OWP;
      c1[j] = x1 + ((j + 1) * sw + OWP - 1) / OWP;
    }
  }
  __syncthreads();
  for (int e = t; e < CCH * OHP * OWP; e += 256) {
    int c = e / (OHP * OWP); int rem = e - c * (OHP * OWP);
    int i = rem / OWP, j = rem - i * OWP;
    const float* IIc = II + (size_t)c * 1089;
    float sum = IIc[r1[i]*33 + c1[j]] - IIc[r0[i]*33 + c1[j]]
              - IIc[r1[i]*33 + c0[j]] + IIc[r0[i]*33 + c0[j]];
    float area = (float)((r1[i] - r0[i]) * (c1[j] - c0[j]));
    rois[((size_t)k * CCH + c) * (OHP * OWP) + rem] = sum / area;
  }
}

// Assemble outputs: boxes(1200) | keep(300, as float) | cls(6300) | bbox(1200)
__global__ void output_kernel(const float* __restrict__ boxes,
                              const int* __restrict__ keep,
                              const float* __restrict__ pred,
                              float* __restrict__ out)
{
  int t = blockIdx.x * blockDim.x + threadIdx.x;
  const int NB = KTOP * 4, NK = KTOP, NC = KTOP * NCLS;
  if (t < NB) {
    out[t] = boxes[t];
  } else if (t < NB + NK) {
    out[t] = keep[t - NB] ? 1.f : 0.f;
  } else if (t < NB + NK + NC) {
    int q = t - (NB + NK); int r = q / NCLS; int c = q - r * NCLS;
    out[t] = pred[r * (NCLS + 4) + c];
  } else if (t < NB + NK + NC + NB) {
    int q = t - (NB + NK + NC); int r = q / 4; int c = q - r * 4;
    out[t] = pred[r * (NCLS + 4) + NCLS + c];
  }
}

// =====================================================================
extern "C" void kernel_launch(void* const* d_in, const int* in_sizes, int n_in,
                              void* d_out, int out_size, void* d_ws, size_t ws_size,
                              hipStream_t stream) {
  (void)in_sizes; (void)n_in; (void)out_size; (void)ws_size;

  // ---- input unpacking (setup_inputs dict order: x, params{...}, anchors) ----
  const float* x       = (const float*)d_in[0];
  const float* bw1     = (const float*)d_in[1];
  const float* bb1     = (const float*)d_in[2];
  const float* bw2     = (const float*)d_in[3];
  const float* bb2     = (const float*)d_in[4];
  const float* bw3     = (const float*)d_in[5];
  const float* bb3     = (const float*)d_in[6];
  const float* rpn_w3  = (const float*)d_in[7];
  const float* rpn_b3  = (const float*)d_in[8];
  const float* rpn_g   = (const float*)d_in[9];
  const float* rpn_be  = (const float*)d_in[10];
  const float* rpn_m   = (const float*)d_in[11];
  const float* rpn_v   = (const float*)d_in[12];
  const float* rpn_w1  = (const float*)d_in[13];
  const float* rpn_b1  = (const float*)d_in[14];
  const float *pw[4], *pb[4], *pg[4], *pbe[4], *pm[4], *pv[4];
  for (int l = 0; l < 4; ++l) {
    int base = 15 + 6 * l;
    pw[l]  = (const float*)d_in[base + 0];
    pb[l]  = (const float*)d_in[base + 1];
    pg[l]  = (const float*)d_in[base + 2];
    pbe[l] = (const float*)d_in[base + 3];
    pm[l]  = (const float*)d_in[base + 4];
    pv[l]  = (const float*)d_in[base + 5];
  }
  const float* fc1_w = (const float*)d_in[39];
  const float* fc1_b = (const float*)d_in[40];
  const float* fc2_w = (const float*)d_in[41];
  const float* fc2_b = (const float*)d_in[42];
  const float* fc3_w = (const float*)d_in[43];
  const float* fc3_b = (const float*)d_in[44];
  const float* anchors = (const float*)d_in[45];
  float* outf = (float*)d_out;

  // ---- workspace carve ----
  float* ws = (float*)d_ws;
  size_t off = 0;
  auto carve = [&](size_t n) { float* p = ws + off; off += (n + 3) & ~(size_t)3; return p; };
  float* f1     = carve((size_t)64 * 128 * 128);
  float* f2     = carve((size_t)256 * 64 * 64);
  float* f3     = carve((size_t)CCH * FHH * FWW);
  float* r3     = carve((size_t)CCH * FHH * FWW);
  float* r1b    = carve((size_t)54 * FHH * FWW);
  float* regb   = carve((size_t)NANCH * 4);
  float* prob   = carve(NANCH);
  float* masked = carve(NANCH);
  float* scores = carve(KTOP);
  int*   idxb   = (int*)carve(KTOP);
  float* boxes  = carve(KTOP * 4);
  int*   valid  = (int*)carve(KTOP);
  float* ious   = carve((size_t)KTOP * KTOP);
  int*   keep   = (int*)carve(KTOP);
  float* II     = carve((size_t)CCH * 33 * 33);
  float* roi0   = carve((size_t)KTOP * CCH * OHP * OWP);
  float* roi1   = carve((size_t)KTOP * CCH * OHP * OWP);
  float* h1     = carve((size_t)KTOP * 4096);
  float* h2     = carve((size_t)KTOP * 4096);
  float* pred   = carve((size_t)KTOP * (NCLS + 4));

  const dim3 wave(32);

  // ---- backbone (K not multiple of 32 for conv1 -> direct-load path) ----
  conv_wmma_kernel<7, 7, 128, false><<<dim3(128 * 128 / 16, 2, 1), wave, 0, stream>>>(
      x, bw1, bb1, nullptr, nullptr, nullptr, nullptr, f1,
      3, 512, 512, 64, 128, 4, 3, 1);
  conv_wmma_kernel<3, 3, 64, true><<<dim3(64 * 64 / 16, 8, 1), wave, 0, stream>>>(
      f1, bw2, bb2, nullptr, nullptr, nullptr, nullptr, f2,
      64, 128, 128, 256, 64, 2, 1, 1);
  conv_wmma_kernel<3, 3, 32, true><<<dim3(32 * 32 / 16, 16, 1), wave, 0, stream>>>(
      f2, bw3, bb3, nullptr, nullptr, nullptr, nullptr, f3,
      256, 64, 64, 512, 32, 2, 1, 1);

  // ---- RPN (TDM weight staging) ----
  conv_wmma_kernel<3, 3, 32, true><<<dim3(64, 16, 1), wave, 0, stream>>>(
      f3, rpn_w3, rpn_b3, rpn_g, rpn_be, rpn_m, rpn_v, r3,
      512, 32, 32, 512, 32, 1, 1, 0);
  conv_wmma_kernel<1, 1, 32, true><<<dim3(64, 2, 1), wave, 0, stream>>>(
      r3, rpn_w1, rpn_b1, nullptr, nullptr, nullptr, nullptr, r1b,
      512, 32, 32, 54, 32, 1, 0, 0);

  rpn_post_kernel<<<(NANCH + 255) / 256, 256, 0, stream>>>(r1b, regb, prob, masked);
  topk_kernel<<<1, 512, 0, stream>>>(masked, scores, idxb);
  decode_kernel<<<(KTOP + 63) / 64, 64, 0, stream>>>(anchors, regb, scores, idxb, boxes, valid);
  iou_kernel<<<(KTOP * KTOP + 255) / 256, 256, 0, stream>>>(boxes, ious);
  nms_kernel<<<1, 512, 0, stream>>>(ious, valid, keep);

  // ---- ROI pooling via integral image ----
  integral_kernel<<<CCH, 64, 0, stream>>>(f3, II);
  roipool_kernel<<<KTOP, 256, 0, stream>>>(II, boxes, roi0);

  // ---- ROI head convs (batched over 300 ROIs, TDM weight staging) ----
  float* src = roi0;
  float* dst = roi1;
  for (int l = 0; l < 4; ++l) {
    conv_wmma_kernel<3, 3, 7, true><<<dim3(4, 16, KTOP), wave, 0, stream>>>(
        src, pw[l], pb[l], pg[l], pbe[l], pm[l], pv[l], dst,
        CCH, OHP, OWP, CCH, OHP, 1, 1, 1);
    float* t = src; src = dst; dst = t;
  }
  // after 4 swaps, `src` holds the final head activations [300][512*49]

  // ---- FC head ----
  gemm_wmma_kernel<<<dim3((KTOP + 15) / 16, 4096 / 32), wave, 0, stream>>>(
      src, fc1_w, fc1_b, h1, KTOP, 4096, CCH * OHP * OWP, 1);
  gemm_wmma_kernel<<<dim3((KTOP + 15) / 16, 4096 / 32), wave, 0, stream>>>(
      h1, fc2_w, fc2_b, h2, KTOP, 4096, 4096, 1);
  gemm_wmma_kernel<<<dim3((KTOP + 15) / 16, 1), wave, 0, stream>>>(
      h2, fc3_w, fc3_b, pred, KTOP, NCLS + 4, 4096, 0);

  // ---- assemble outputs ----
  const int total = KTOP * 4 + KTOP + KTOP * NCLS + KTOP * 4;
  output_kernel<<<(total + 255) / 256, 256, 0, stream>>>(boxes, keep, pred, outf);
}
